// CausalSelfAttention_40810779247124
// MI455X (gfx1250) — matrix operand order repaired and checked by reading, not compiled, not run on hardware
//
#include <hip/hip_runtime.h>
#include <hip/hip_bf16.h>
#include <math.h>

typedef __bf16 bf16;
typedef __attribute__((ext_vector_type(16))) __bf16 v16bf;
typedef __attribute__((ext_vector_type(8)))  __bf16 v8bf;
typedef __attribute__((ext_vector_type(8)))  float  v8f;
typedef __attribute__((ext_vector_type(4)))  float  v4f;
typedef __attribute__((ext_vector_type(4)))  int    v4i;

#define N_STATE 2048
#define N_HEADS 16
#define D_HEAD  128
#define BATCH   2
#define SEQ     2048
#define MROWS   (BATCH * SEQ)   // 4096

#if __has_builtin(__builtin_amdgcn_global_load_async_to_lds_b128)
#define HAVE_ASYNC_LDS 1
typedef v4i* gas_v4i;                                      // global src (v4i*)
typedef __attribute__((address_space(3))) v4i* lds_v4i;    // LDS dst
#else
#define HAVE_ASYNC_LDS 0
#endif

static __device__ __forceinline__ void wait_async0() {
#if __has_builtin(__builtin_amdgcn_s_wait_asynccnt)
  __builtin_amdgcn_s_wait_asynccnt(0);
#else
  asm volatile("s_wait_asynccnt 0x0" ::: "memory");
#endif
}

// ---------------------------------------------------------------------------
// helpers
// ---------------------------------------------------------------------------
static __device__ __forceinline__ v16bf make16(v8bf lo, v8bf hi) {
  v16bf r;
#pragma unroll
  for (int i = 0; i < 8; ++i) { r[i] = lo[i]; r[i + 8] = hi[i]; }
  return r;
}

// A-matrix fragment (16 x 32 bf16), ISA layout:
// lanes 0-15: M=lane, K = koff..koff+7 then koff+16..koff+23 with koff=0
// lanes 16-31: same rows, koff=8. Source is row-major [rows][ld] (K contiguous).
static __device__ __forceinline__ v16bf load_a_frag(const bf16* __restrict__ src,
                                                    int ld, int row0, int k0) {
  const int lane = threadIdx.x & 31;
  const bf16* p = src + (size_t)(row0 + (lane & 15)) * ld + k0 + ((lane >> 4) << 3);
  v8bf lo = *(const v8bf*)p;
  v8bf hi = *(const v8bf*)(p + 16);
  return make16(lo, hi);
}

// B-matrix fragment (32 x 16 bf16) where column n = row n of a row-major
// K-contiguous source (i.e. computes X * W^T):
// lanes 0-15: N=lane, K=0..7,16..23 ; lanes 16-31: N=lane-16, K=8..15,24..31
static __device__ __forceinline__ v16bf load_b_frag(const bf16* __restrict__ src,
                                                    int ld, int col0, int k0) {
  const int lane = threadIdx.x & 31;
  const bf16* p = src + (size_t)(col0 + (lane & 15)) * ld + k0 + ((lane >> 4) << 3);
  v8bf lo = *(const v8bf*)p;
  v8bf hi = *(const v8bf*)(p + 16);
  return make16(lo, hi);
}

static __device__ __forceinline__ v8f wmma_bf16(v16bf a, v16bf b, v8f c) {
  return __builtin_amdgcn_wmma_f32_16x16x32_bf16(false, a, false, b, (short)0, c,
                                                 false, false);
}

// ---------------------------------------------------------------------------
// fp32 -> bf16 conversion (8 elements / thread, vectorized)
// ---------------------------------------------------------------------------
__global__ void cvt_f32_bf16(const float* __restrict__ x, bf16* __restrict__ y, int n) {
  int i = (blockIdx.x * blockDim.x + threadIdx.x) * 8;
  if (i >= n) return;
  v4f a = *(const v4f*)(x + i);
  v4f b = *(const v4f*)(x + i + 4);
  v8bf o;
#pragma unroll
  for (int j = 0; j < 4; ++j) { o[j] = (bf16)a[j]; o[j + 4] = (bf16)b[j]; }
  *(v8bf*)(y + i) = o;
}

// ---------------------------------------------------------------------------
// WMMA GEMM: C[M,N] = A[M,K] * B[N,K]^T   (all bf16, fp32 accumulate)
// block = 128 threads (4 waves, 2x2), wave tile 64x64, block tile 128x128
// MODE 0: scatter into Q/K/V head-major bf16 buffers (QKV projection)
// MODE 1: store fp32 to Out (output projection)
// ---------------------------------------------------------------------------
template <int MODE>
__global__ void __launch_bounds__(128)
gemm_bf16(const bf16* __restrict__ A, const bf16* __restrict__ Bm,
          int M, int N, int K,
          bf16* __restrict__ Qh, bf16* __restrict__ Kh, bf16* __restrict__ Vh,
          float* __restrict__ Out) {
  const int lane = threadIdx.x & 31;
  const int wave = threadIdx.x >> 5;
  const int wm = wave >> 1, wn = wave & 1;
  const int m0 = blockIdx.y * 128 + wm * 64;
  const int n0 = blockIdx.x * 128 + wn * 64;

  v8f acc[4][4];
#pragma unroll
  for (int im = 0; im < 4; ++im)
#pragma unroll
    for (int in = 0; in < 4; ++in)
#pragma unroll
      for (int r = 0; r < 8; ++r) acc[im][in][r] = 0.0f;

  for (int k0 = 0; k0 < K; k0 += 32) {
    if (k0 + 32 < K) {  // prefetch next K tile (global_prefetch_b8)
      __builtin_prefetch(A + (size_t)(m0 + (lane & 15)) * K + k0 + 32, 0, 1);
      __builtin_prefetch(Bm + (size_t)(n0 + (lane & 15)) * K + k0 + 32, 0, 1);
    }
    v16bf Af[4];
#pragma unroll
    for (int im = 0; im < 4; ++im) Af[im] = load_a_frag(A, K, m0 + 16 * im, k0);
#pragma unroll
    for (int in = 0; in < 4; ++in) {
      v16bf Bf = load_b_frag(Bm, K, n0 + 16 * in, k0);
#pragma unroll
      for (int im = 0; im < 4; ++im) acc[im][in] = wmma_bf16(Af[im], Bf, acc[im][in]);
    }
  }

  // C/D layout: lanes 0-15 col N=lane, row r ; lanes 16-31 col N=lane-16, row r+8
  const int half = lane >> 4;
  const int colL = lane & 15;
#pragma unroll
  for (int im = 0; im < 4; ++im)
#pragma unroll
    for (int in = 0; in < 4; ++in)
#pragma unroll
      for (int r = 0; r < 8; ++r) {
        const int m = m0 + im * 16 + r + 8 * half;
        const int n = n0 + in * 16 + colL;
        const float v = acc[im][in][r];
        if (MODE == 0) {
          const int sec = n >> 11;            // 0=Q 1=K 2=V
          const int inner = n & 2047;
          const int hh = inner >> 7, dd = inner & 127;
          const int bb = m >> 11, ss = m & 2047;
          const size_t idx =
              (((size_t)(bb * N_HEADS + hh)) * SEQ + ss) * D_HEAD + dd;
          const bf16 bv = (bf16)v;
          if (sec == 0) Qh[idx] = bv;
          else if (sec == 1) Kh[idx] = bv;
          else Vh[idx] = bv;
        } else {
          Out[(size_t)m * N + n] = v;
        }
      }
}

// ---------------------------------------------------------------------------
// Flash attention (causal) over head-major bf16 Q/K/V [B,H,S,128].
// block = 128 threads (4 waves); wave owns 16 query rows; 32-key tiles.
// V tiles are double-buffered into LDS via async LDS-DMA (ASYNCcnt), then
// transposed LDS->LDS into a padded V^T buffer feeding the PV B-fragments.
// Output: bf16 [B*S, 2048] (row-major, head-concatenated) for the out-proj.
// ---------------------------------------------------------------------------
#define VTS 40  // padded LDS stride (bf16) for V^T rows -> 80B, b128 aligned
#define PS  40  // padded LDS stride (bf16) for P rows

__global__ void __launch_bounds__(128)
flash_attn(const bf16* __restrict__ Qb, const bf16* __restrict__ Kb,
           const bf16* __restrict__ Vb, bf16* __restrict__ Ob) {
  __shared__ bf16 Vstage[2][32 * D_HEAD];  // row-major V tiles (async dst)
  __shared__ bf16 Vt[D_HEAD * VTS];        // V^T tile: [d][k], 32 keys
  __shared__ bf16 Pl[4 * 16 * PS];         // per-wave P staging (16x32)

  const int lane = threadIdx.x & 31;
  const int wave = threadIdx.x >> 5;
  const int bx = blockIdx.x;               // 64-row query block
  const int h = blockIdx.y;
  const int b = blockIdx.z;

  const size_t headoff = ((size_t)(b * N_HEADS + h)) * SEQ * D_HEAD;
  const bf16* Qh = Qb + headoff;
  const bf16* Kh = Kb + headoff;
  const bf16* Vh = Vb + headoff;

  const int q0 = bx * 64 + wave * 16;

  // Q fragments for the whole K-dim (128 = 4 x 32)
  v16bf Qf[4];
#pragma unroll
  for (int kk = 0; kk < 4; ++kk) Qf[kk] = load_a_frag(Qh, D_HEAD, q0, kk * 32);

  v8f o[8];
#pragma unroll
  for (int dc = 0; dc < 8; ++dc)
#pragma unroll
    for (int r = 0; r < 8; ++r) o[dc][r] = 0.0f;
  float mi[8], li[8];
#pragma unroll
  for (int r = 0; r < 8; ++r) { mi[r] = -3.0e38f; li[r] = 0.0f; }

  const float scale = 0.08838834764831845f;  // 1/sqrt(128)
  const int half = lane >> 4;
  const int colL = lane & 15;
  const int jcount = bx * 2 + 2;             // key tiles needed by this block
  bf16* Pw = Pl + wave * 16 * PS;

#if HAVE_ASYNC_LDS
  // prologue: async-stage V tile 0 (32 keys x 128 dims, 16B per lane-chunk)
#pragma unroll
  for (int it = 0; it < 4; ++it) {
    const int c = threadIdx.x + 128 * it;
    const int row = c >> 4;
    const int cc = (c & 15) << 3;
    __builtin_amdgcn_global_load_async_to_lds_b128(
        (gas_v4i)(Vh + (size_t)row * D_HEAD + cc),
        (lds_v4i)(&Vstage[0][row * D_HEAD + cc]), 0, 0);
  }
#endif

  for (int j = 0; j < jcount; ++j) {
#if HAVE_ASYNC_LDS
    wait_async0();                       // this wave's stage writes landed
    __syncthreads();                     // all waves' writes visible; Vt free
    // ---- LDS->LDS transpose of staged tile into padded V^T ----
    {
      const bf16* st = Vstage[j & 1];
#pragma unroll
      for (int it = 0; it < 4; ++it) {
        const int c = threadIdx.x + 128 * it;
        const int row = c >> 4;
        const int cc = (c & 15) << 3;
        v8bf vv = *(const v8bf*)(st + row * D_HEAD + cc);
#pragma unroll
        for (int e = 0; e < 8; ++e) Vt[(cc + e) * VTS + row] = vv[e];
      }
    }
    __syncthreads();                     // V^T ready
    if (j + 1 < jcount) {                // async prefetch of next tile
      bf16* st = Vstage[(j + 1) & 1];
#pragma unroll
      for (int it = 0; it < 4; ++it) {
        const int c = threadIdx.x + 128 * it;
        const int row = c >> 4;
        const int cc = (c & 15) << 3;
        __builtin_amdgcn_global_load_async_to_lds_b128(
            (gas_v4i)(Vh + (size_t)((j + 1) * 32 + row) * D_HEAD + cc),
            (lds_v4i)(st + row * D_HEAD + cc), 0, 0);
      }
    }
#else
    // fallback: synchronous cooperative V^T staging
    __syncthreads();
#pragma unroll
    for (int it = 0; it < 4; ++it) {
      const int c = threadIdx.x + 128 * it;
      const int row = c >> 4;
      const int cc = (c & 15) << 3;
      v8bf vv = *(const v8bf*)(Vh + (size_t)(j * 32 + row) * D_HEAD + cc);
#pragma unroll
      for (int e = 0; e < 8; ++e) Vt[(cc + e) * VTS + row] = vv[e];
    }
    __syncthreads();
#endif

    if (j * 32 <= q0 + 15) {                      // causal: wave has live keys
      // ---- scores S = Q K^T (two 16x16 tiles -> 16x32) ----
      v8f sacc[2];
#pragma unroll
      for (int t = 0; t < 2; ++t)
#pragma unroll
        for (int r = 0; r < 8; ++r) sacc[t][r] = 0.0f;
#pragma unroll
      for (int kk = 0; kk < 4; ++kk)
#pragma unroll
        for (int t = 0; t < 2; ++t) {
          v16bf Bf = load_b_frag(Kh, D_HEAD, j * 32 + t * 16, kk * 32);
          sacc[t] = wmma_bf16(Qf[kk], Bf, sacc[t]);
        }
      // ---- scale + causal mask ----
#pragma unroll
      for (int t = 0; t < 2; ++t)
#pragma unroll
        for (int r = 0; r < 8; ++r) {
          const int key = j * 32 + t * 16 + colL;
          const int qrow = q0 + r + 8 * half;
          const float s = sacc[t][r] * scale;
          sacc[t][r] = (key <= qrow) ? s : -3.0e38f;
        }
      // ---- online softmax: row max over 16-lane halves ----
      float mnew[8], alpha[8];
#pragma unroll
      for (int r = 0; r < 8; ++r) {
        float rm = fmaxf(sacc[0][r], sacc[1][r]);
#pragma unroll
        for (int d = 8; d >= 1; d >>= 1) rm = fmaxf(rm, __shfl_xor(rm, d, 32));
        mnew[r] = fmaxf(mi[r], rm);
        alpha[r] = __expf(mi[r] - mnew[r]);
        mi[r] = mnew[r];
      }
      // ---- p = exp(s - m), row sums ----
      float rs[8];
#pragma unroll
      for (int r = 0; r < 8; ++r) rs[r] = 0.0f;
#pragma unroll
      for (int t = 0; t < 2; ++t)
#pragma unroll
        for (int r = 0; r < 8; ++r) {
          const float p = __expf(sacc[t][r] - mnew[r]);
          sacc[t][r] = p;
          rs[r] += p;
        }
#pragma unroll
      for (int r = 0; r < 8; ++r) {
        float v = rs[r];
#pragma unroll
        for (int d = 8; d >= 1; d >>= 1) v += __shfl_xor(v, d, 32);
        li[r] = li[r] * alpha[r] + v;
      }
      // ---- rescale running O ----
#pragma unroll
      for (int dc = 0; dc < 8; ++dc)
#pragma unroll
        for (int r = 0; r < 8; ++r) o[dc][r] *= alpha[r];
      // ---- P: C-layout -> A-layout via per-wave LDS (intra-wave, DS in-order) ----
#pragma unroll
      for (int t = 0; t < 2; ++t)
#pragma unroll
        for (int r = 0; r < 8; ++r)
          Pw[(r + 8 * half) * PS + t * 16 + colL] = (bf16)sacc[t][r];
      __builtin_amdgcn_wave_barrier();
      asm volatile("s_wait_dscnt 0x0" ::: "memory");
      {
        const bf16* pp = Pw + (size_t)(lane & 15) * PS + ((lane >> 4) << 3);
        v8bf plo = *(const v8bf*)pp;
        v8bf phi = *(const v8bf*)(pp + 16);
        v16bf Pf = make16(plo, phi);
        // ---- O += P * V  (B-frag columns are contiguous rows of V^T) ----
#pragma unroll
        for (int dc = 0; dc < 8; ++dc) {
          const bf16* vp = Vt + (size_t)(dc * 16 + colL) * VTS + (half << 3);
          v8bf blo = *(const v8bf*)vp;
          v8bf bhi = *(const v8bf*)(vp + 16);
          v16bf Bf = make16(blo, bhi);
          o[dc] = wmma_bf16(Pf, Bf, o[dc]);
        }
      }
    }
  }

  // ---- epilogue: O / l, store bf16 head-concatenated [B*S, 2048] ----
  float rinv[8];
#pragma unroll
  for (int r = 0; r < 8; ++r) rinv[r] = 1.0f / li[r];
#pragma unroll
  for (int dc = 0; dc < 8; ++dc)
#pragma unroll
    for (int r = 0; r < 8; ++r) {
      const int srow = q0 + r + 8 * half;
      const int d = dc * 16 + colL;
      Ob[((size_t)(b * SEQ + srow)) * N_STATE + h * D_HEAD + d] =
          (bf16)(o[dc][r] * rinv[r]);
    }
}

// ---------------------------------------------------------------------------
// launch
// ---------------------------------------------------------------------------
extern "C" void kernel_launch(void* const* d_in, const int* in_sizes, int n_in,
                              void* d_out, int out_size, void* d_ws, size_t ws_size,
                              hipStream_t stream) {
  (void)in_sizes; (void)n_in; (void)out_size; (void)ws_size;
  const float* x    = (const float*)d_in[0];  // [B,S,2048]
  const float* wqkv = (const float*)d_in[3];  // [6144,2048]
  const float* wout = (const float*)d_in[4];  // [2048,2048]

  char* ws = (char*)d_ws;
  bf16* Xb = (bf16*)ws;  ws += (size_t)MROWS * N_STATE * 2;
  bf16* Wq = (bf16*)ws;  ws += (size_t)3 * N_STATE * N_STATE * 2;
  bf16* Wo = (bf16*)ws;  ws += (size_t)N_STATE * N_STATE * 2;
  bf16* Qb = (bf16*)ws;  ws += (size_t)MROWS * N_STATE * 2;
  bf16* Kb = (bf16*)ws;  ws += (size_t)MROWS * N_STATE * 2;
  bf16* Vb = (bf16*)ws;  ws += (size_t)MROWS * N_STATE * 2;
  bf16* Ab = (bf16*)ws;  ws += (size_t)MROWS * N_STATE * 2;

  const int nX = MROWS * N_STATE;            // 8,388,608
  const int nWq = 3 * N_STATE * N_STATE;     // 12,582,912
  const int nWo = N_STATE * N_STATE;         // 4,194,304
  cvt_f32_bf16<<<(nX / 8 + 255) / 256, 256, 0, stream>>>(x, Xb, nX);
  cvt_f32_bf16<<<(nWq / 8 + 255) / 256, 256, 0, stream>>>(wqkv, Wq, nWq);
  cvt_f32_bf16<<<(nWo / 8 + 255) / 256, 256, 0, stream>>>(wout, Wo, nWo);

  // QKV projection: [4096,2048] x [6144,2048]^T, scatter into head-major Q/K/V
  gemm_bf16<0><<<dim3(3 * N_STATE / 128, MROWS / 128), 128, 0, stream>>>(
      Xb, Wq, MROWS, 3 * N_STATE, N_STATE, Qb, Kb, Vb, nullptr);

  // causal flash attention per (64-row block, head, batch)
  flash_attn<<<dim3(SEQ / 64, N_HEADS, BATCH), 128, 0, stream>>>(Qb, Kb, Vb, Ab);

  // output projection: [4096,2048] x [2048,2048]^T -> fp32 d_out
  gemm_bf16<1><<<dim3(N_STATE / 128, MROWS / 128), 128, 0, stream>>>(
      Ab, Wo, MROWS, N_STATE, N_STATE, nullptr, nullptr, nullptr, (float*)d_out);
}
